// SoftLogicEngine_18468359372881
// MI455X (gfx1250) — matrix operand order repaired and checked by reading, not compiled, not run on hardware
//
#include <hip/hip_runtime.h>
#include <hip/hip_bf16.h>

typedef __attribute__((ext_vector_type(16))) _Float16 v16h;
typedef __attribute__((ext_vector_type(8)))  _Float16 v8h;
typedef __attribute__((ext_vector_type(8)))  float    v8f;

#define TAU 0.05f
#define BETA 8.0f
#define MAX_ITERS 6
#define EPSN 1e-12f
#define DDIM 128
#define HSLOT 4
#define TILES_PER_BLOCK 50

// ---------- helpers ----------
__device__ __forceinline__ unsigned sortable_key(float f) {
  unsigned b = __float_as_uint(f);
  return (b & 0x80000000u) ? ~b : (b | 0x80000000u);
}
__device__ __forceinline__ float unsortable_key(unsigned key) {
  unsigned b = (key & 0x80000000u) ? (key ^ 0x80000000u) : ~key;
  return __uint_as_float(b);
}

// ---------- 1. normalize K rows -> f16 A [P,128] ----------
__global__ __launch_bounds__(128) void norm_k_kernel(const float* __restrict__ K,
                                                     _Float16* __restrict__ A) {
  int row = blockIdx.x;
  int t = threadIdx.x;
  float x = K[row * DDIM + t];
  float s = x * x;
  #pragma unroll
  for (int off = 16; off > 0; off >>= 1) s += __shfl_down(s, off, 32);
  __shared__ float red[4];
  if ((t & 31) == 0) red[t >> 5] = s;
  __syncthreads();
  float tot = red[0] + red[1] + red[2] + red[3];
  float denom = fmaxf(sqrtf(tot), EPSN);
  A[row * DDIM + t] = (_Float16)(x / denom);
}

// ---------- 2. normalize Theta columns (over d) -> f16 Bt [C*h, 128] ----------
__global__ __launch_bounds__(128) void norm_theta_kernel(const float* __restrict__ Theta,
                                                         _Float16* __restrict__ Bt) {
  int n = blockIdx.x;          // n = c*h + s
  int k = threadIdx.x;         // d index
  int c = n >> 2;
  int s = n & 3;
  float x = Theta[(c * DDIM + k) * HSLOT + s];
  float sq = x * x;
  #pragma unroll
  for (int off = 16; off > 0; off >>= 1) sq += __shfl_down(sq, off, 32);
  __shared__ float red[4];
  if ((k & 31) == 0) red[k >> 5] = sq;
  __syncthreads();
  float tot = red[0] + red[1] + red[2] + red[3];
  float denom = fmaxf(sqrtf(tot), EPSN);
  Bt[n * DDIM + k] = (_Float16)(x / denom);
}

// ---------- 3. init packed max array ----------
__global__ void init_packed_kernel(unsigned long long* __restrict__ packed, int n) {
  int i = blockIdx.x * blockDim.x + threadIdx.x;
  if (i < n) packed[i] = 0ull;
}

// ---------- 4. fused WMMA GEMM + column max/argmax ----------
__device__ __forceinline__ v16h load_a_frag(const _Float16* __restrict__ arow, int kb, int hk) {
  // 16-bit A 16x32 layout: halves 0..15 -> K = kb+hk+0..15 (contiguous, matches B)
  v8h lo = *(const v8h*)(arow + kb + hk);
  v8h hi = *(const v8h*)(arow + kb + hk + 8);
  v16h r;
  #pragma unroll
  for (int i = 0; i < 8; ++i) { r[i] = lo[i]; r[i + 8] = hi[i]; }
  return r;
}

__global__ __launch_bounds__(128) void gemm_max_kernel(const _Float16* __restrict__ A,
                                                       const _Float16* __restrict__ Bt,
                                                       unsigned long long* __restrict__ packed,
                                                       int P) {
  const int wave = threadIdx.x >> 5;
  const int lane = threadIdx.x & 31;
  const int l15 = lane & 15;
  const int hk = (lane >> 4) * 16;   // A fragment K sub-offset
  const int hi16 = (lane >> 4) * 16; // B fragment K sub-offset

  // B fragments for one 16-column tile, all of d=128 (4 chunks of K=32)
  const int n = (blockIdx.x * 4 + wave) * 16 + l15;
  const _Float16* __restrict__ brow = Bt + (long)n * DDIM;
  v16h b0 = *(const v16h*)(brow + 0  + hi16);
  v16h b1 = *(const v16h*)(brow + 32 + hi16);
  v16h b2 = *(const v16h*)(brow + 64 + hi16);
  v16h b3 = *(const v16h*)(brow + 96 + hi16);

  const int numRowTiles = P >> 4;
  const int rowTile0 = blockIdx.y * TILES_PER_BLOCK;
  int tCount = numRowTiles - rowTile0;
  if (tCount > TILES_PER_BLOCK) tCount = TILES_PER_BLOCK; // block-uniform

  const _Float16* __restrict__ arow = A + (long)(rowTile0 * 16 + l15) * DDIM;

  float bestV = -3.0e38f;
  int bestI = 0;
  for (int t = 0; t < tCount; ++t) {
    v16h a0 = load_a_frag(arow, 0, hk);
    v16h a1 = load_a_frag(arow, 32, hk);
    v16h a2 = load_a_frag(arow, 64, hk);
    v16h a3 = load_a_frag(arow, 96, hk);
    v8f acc = {};
    acc = __builtin_amdgcn_wmma_f32_16x16x32_f16(false, a0, false, b0, (short)0, acc, false, false);
    acc = __builtin_amdgcn_wmma_f32_16x16x32_f16(false, a1, false, b1, (short)0, acc, false, false);
    acc = __builtin_amdgcn_wmma_f32_16x16x32_f16(false, a2, false, b2, (short)0, acc, false, false);
    acc = __builtin_amdgcn_wmma_f32_16x16x32_f16(false, a3, false, b3, (short)0, acc, false, false);
    // C layout: lanes 0-15: VGPR i -> row M=i; lanes 16-31: VGPR i -> row M=8+i; col N=lane&15
    int pbase = (rowTile0 + t) * 16 + (lane >> 4) * 8;
    #pragma unroll
    for (int i = 0; i < 8; ++i) {
      float v = acc[i];
      if (v > bestV) { bestV = v; bestI = pbase + i; }  // strict > keeps lowest p
    }
    arow += 16 * DDIM;
  }

  // pack (sortable value, ~index): max over packed == argmax with lowest-index tiebreak
  unsigned key = sortable_key(bestV);
  unsigned lo = ~(unsigned)bestI;
  unsigned olo = __shfl_xor(lo, 16, 32);
  unsigned okey = __shfl_xor(key, 16, 32);
  unsigned long long pk = ((unsigned long long)key << 32) | lo;
  unsigned long long opk = ((unsigned long long)okey << 32) | olo;
  if (opk > pk) pk = opk;
  if (lane < 16) atomicMax(&packed[n], pk);
}

// ---------- 5. unpack best/conf ----------
__global__ void finalize_kernel(const unsigned long long* __restrict__ packed,
                                float* __restrict__ conf, int* __restrict__ best, int n) {
  int i = blockIdx.x * blockDim.x + threadIdx.x;
  if (i >= n) return;
  unsigned long long pk = packed[i];
  unsigned key = (unsigned)(pk >> 32);
  conf[i] = unsortable_key(key);
  best[i] = (int)(~(unsigned)(pk & 0xffffffffull));
}

// ---------- 6. copy initial_val -> val (d_out) ----------
__global__ void copy_init_kernel(const float* __restrict__ src, float* __restrict__ dst, int n) {
  int i = blockIdx.x * blockDim.x + threadIdx.x;
  if (i < n) dst[i] = src[i];
}

// ---------- 7. per-clause softmin -> exp(beta*g) ----------
__global__ void clause_kernel(const float* __restrict__ conf, const int* __restrict__ best,
                              const float* __restrict__ val, float* __restrict__ gexp, int C) {
  int c = blockIdx.x * blockDim.x + threadIdx.x;
  if (c >= C) return;
  float x[2 * HSLOT];
  #pragma unroll
  for (int s = 0; s < HSLOT; ++s) {
    x[s] = conf[c * HSLOT + s];
    x[HSLOT + s] = val[best[c * HSLOT + s]];
  }
  // g = -(1/TAU) * (logsumexp(-x/TAU) - log(2h)), clipped to [0,1]
  float mn = x[0];
  #pragma unroll
  for (int i = 1; i < 2 * HSLOT; ++i) mn = fminf(mn, x[i]);
  float m = -mn / TAU; // max of -x/TAU
  float sum = 0.0f;
  #pragma unroll
  for (int i = 0; i < 2 * HSLOT; ++i) sum += __expf(-x[i] / TAU - m);
  float lse = m + __logf(sum);
  float g = -(1.0f / TAU) * (lse - __logf((float)(2 * HSLOT)));
  g = fminf(fmaxf(g, 0.0f), 1.0f);
  gexp[c] = __expf(BETA * g);
}

// ---------- 8. per-head soft-OR update (deterministic, fixed clause order) ----------
__global__ void head_update_kernel(const float* __restrict__ gexp,
                                   const int* __restrict__ head_idx,
                                   float* __restrict__ val, int P, int C) {
  int p = blockIdx.x * blockDim.x + threadIdx.x;
  if (p >= P) return;
  float e = 0.0f;
  for (int c = 0; c < C; ++c) {
    if (head_idx[c] == p) e += gexp[c]; // in-order sum == segment_sum ordering
  }
  if (e > 0.0f) {
    float gh = __logf(fmaxf(e, 1e-38f)) / BETA;
    val[p] = fmaxf(val[p], gh);
  }
}

extern "C" void kernel_launch(void* const* d_in, const int* in_sizes, int n_in,
                              void* d_out, int out_size, void* d_ws, size_t ws_size,
                              hipStream_t stream) {
  const float* K           = (const float*)d_in[0];   // [P, 128]
  const float* Theta       = (const float*)d_in[1];   // [C, 128, 4]
  const int*   head_idx    = (const int*)d_in[2];     // [C]
  const float* initial_val = (const float*)d_in[3];   // [P]
  float* val = (float*)d_out;                         // [P]

  const int C = in_sizes[2];
  const int P = in_sizes[3];
  const int N = C * HSLOT; // gemm columns

  // workspace layout (256B-aligned sections)
  char* ws = (char*)d_ws;
  size_t off = 0;
  _Float16* A = (_Float16*)(ws + off);              off += (size_t)P * DDIM * sizeof(_Float16);
  off = (off + 255) & ~(size_t)255;
  _Float16* Bt = (_Float16*)(ws + off);             off += (size_t)N * DDIM * sizeof(_Float16);
  off = (off + 255) & ~(size_t)255;
  unsigned long long* packed = (unsigned long long*)(ws + off); off += (size_t)N * 8;
  off = (off + 255) & ~(size_t)255;
  float* conf = (float*)(ws + off);                 off += (size_t)N * 4;
  off = (off + 255) & ~(size_t)255;
  int* best = (int*)(ws + off);                     off += (size_t)N * 4;
  off = (off + 255) & ~(size_t)255;
  float* gexp = (float*)(ws + off);                 off += (size_t)C * 4;
  (void)ws_size;

  // 1-2: normalize & pack to f16
  norm_k_kernel<<<P, 128, 0, stream>>>(K, A);
  norm_theta_kernel<<<N, 128, 0, stream>>>(Theta, Bt);

  // 3: init packed reduction targets
  init_packed_kernel<<<(N + 255) / 256, 256, 0, stream>>>(packed, N);

  // 4: fused WMMA GEMM + column max/argmax
  int numRowTiles = P >> 4; // P multiple of 16
  dim3 gemmGrid(N / 64, (numRowTiles + TILES_PER_BLOCK - 1) / TILES_PER_BLOCK);
  gemm_max_kernel<<<gemmGrid, 128, 0, stream>>>(A, Bt, packed, P);

  // 5: unpack
  finalize_kernel<<<(N + 255) / 256, 256, 0, stream>>>(packed, conf, best, N);

  // 6: val = initial_val
  copy_init_kernel<<<(P + 255) / 256, 256, 0, stream>>>(initial_val, val, P);

  // 7-8: fixed-point iterations
  for (int it = 0; it < MAX_ITERS; ++it) {
    clause_kernel<<<(C + 255) / 256, 256, 0, stream>>>(conf, best, val, gexp, C);
    head_update_kernel<<<(P + 255) / 256, 256, 0, stream>>>(gexp, head_idx, val, P, C);
  }
}
